// EncoderLayer_55181739819371
// MI455X (gfx1250) — matrix operand order repaired and checked by reading, not compiled, not run on hardware
//
#include <hip/hip_runtime.h>
#include <hip/hip_bf16.h>

// ---------------------------------------------------------------------------
// EncoderLayer for MI455X (gfx1250): bf16 WMMA GEMMs + fused banded attention
// B=2, S=2048, D=1024, H=16, DK=DV=64, DFF=4096, WIN=128
// ---------------------------------------------------------------------------

#define BB_   2
#define S_    2048
#define D_    1024
#define H_    16
#define DK_   64
#define DV_   64
#define DFF_  4096
#define WIN_  128
#define NTOK_ (BB_ * S_)   // 4096

typedef __attribute__((ext_vector_type(16))) __bf16 v16bf;
typedef __attribute__((ext_vector_type(8)))  float  v8f;

__device__ __forceinline__ unsigned short f2bf_u(float f) {
  unsigned u = __builtin_bit_cast(unsigned, f);
  if ((u & 0x7fffffffu) > 0x7f800000u) return (unsigned short)0x7fc0u; // NaN
  unsigned r = u + 0x7fffu + ((u >> 16) & 1u);                         // RNE
  return (unsigned short)(r >> 16);
}
__device__ __forceinline__ __bf16 f2bf(float f) {
  unsigned short s = f2bf_u(f);
  return __builtin_bit_cast(__bf16, s);
}

union FragAB {
  v16bf v;
  uint4 q[2];
  __bf16 e[16];
};

__device__ __forceinline__ v8f wmma_bf16(v16bf a, v16bf b, v8f c) {
  // D = A(16x32 bf16) * B(32x16 bf16) + C(16x16 f32)
  return __builtin_amdgcn_wmma_f32_16x16x32_bf16(
      /*neg_a=*/false, a, /*neg_b=*/false, b,
      /*c_mod=*/(short)0, c, /*reuse_a=*/false, /*reuse_b=*/false);
}

// ------------------------------ converters ---------------------------------

__global__ void k_cvt_bf16(const float* __restrict__ in, __bf16* __restrict__ out, int n) {
  int i = blockIdx.x * blockDim.x + threadIdx.x;
  if (i < n) out[i] = f2bf(in[i]);
}

// W[K][N] (row-major fp32) -> Wt[N][K] (row-major bf16)
__global__ void k_cvt_bf16_T(const float* __restrict__ W, __bf16* __restrict__ Wt,
                             int K, int N) {
  int i = blockIdx.x * blockDim.x + threadIdx.x;
  if (i < K * N) {
    int k = i / N, n = i - k * N;
    Wt[(size_t)n * K + k] = f2bf(W[i]);
  }
}

// ------------------------------ WMMA GEMM ----------------------------------
// C[M,N] = alpha * A[M,K] @ Bt[N,K]^T + bias[N]   (batched via blockIdx.z)
// Block tile 128x64, 8 waves as 4x2, each wave 32x32 (2x2 WMMA of 16x16x32).

template<typename AT, bool OUT_BF16, bool OUT_TRANS, bool RELU>
__global__ __launch_bounds__(256) void k_gemm(
    const AT* __restrict__ A, const __bf16* __restrict__ Bt,
    const float* __restrict__ bias, void* __restrict__ Cp,
    int M, int N, int K, int lda, int ldb, int ldc,
    long long aStrB, long long aStrH, long long bStrB, long long bStrH,
    long long cStrB, long long cStrH, int nH, float alpha)
{
  int z  = blockIdx.z;
  int zb = z / nH, zh = z - zb * nH;
  A  += zb * aStrB + zh * aStrH;
  Bt += zb * bStrB + zh * bStrH;
  long long cOff = zb * cStrB + zh * cStrH;

  __shared__ __align__(16) __bf16 sA[128 * 32];
  __shared__ __align__(16) __bf16 sB[64 * 32];

  int t    = threadIdx.x;
  int lane = t & 31;
  int wid  = t >> 5;
  int wm   = wid >> 1;        // 0..3  (M direction)
  int wn   = wid & 1;         // 0..1  (N direction)
  int lo   = lane & 15;
  int hi   = lane >> 4;

  int bm = blockIdx.y * 128;
  int bn = blockIdx.x * 64;

  v8f acc[2][2] = {};

  int arow = t >> 2;          // 0..63
  int acol = (t & 3) * 8;     // 0,8,16,24

  for (int k0 = 0; k0 < K; k0 += 32) {
    // ---- stage A tile (128x32) ----
#pragma unroll
    for (int i = 0; i < 2; ++i) {
      int r = arow + i * 64;
      long long goff = (long long)(bm + r) * lda + k0 + acol;
      uint4 u;
      if constexpr (sizeof(AT) == 2) {
        u = *(const uint4*)(A + goff);
      } else {
        float4 f0 = *(const float4*)(A + goff);
        float4 f1 = *(const float4*)(A + goff + 4);
        u.x = (unsigned)f2bf_u(f0.x) | ((unsigned)f2bf_u(f0.y) << 16);
        u.y = (unsigned)f2bf_u(f0.z) | ((unsigned)f2bf_u(f0.w) << 16);
        u.z = (unsigned)f2bf_u(f1.x) | ((unsigned)f2bf_u(f1.y) << 16);
        u.w = (unsigned)f2bf_u(f1.z) | ((unsigned)f2bf_u(f1.w) << 16);
      }
      *(uint4*)(&sA[r * 32 + acol]) = u;
    }
    // ---- stage Bt tile (64 rows of N x 32 of K) ----
    {
      long long goff = (long long)(bn + arow) * ldb + k0 + acol;
      *(uint4*)(&sB[arow * 32 + acol]) = *(const uint4*)(Bt + goff);
    }
    __syncthreads();

    // ---- fragments per ISA layouts ----
    FragAB a[2], b[2];
#pragma unroll
    for (int ms = 0; ms < 2; ++ms) {
      int m  = wm * 32 + ms * 16 + lo;
      int kb = hi * 8;  // low lanes K 0-7 & 16-23, high lanes 8-15 & 24-31
      a[ms].q[0] = *(const uint4*)(&sA[m * 32 + kb]);
      a[ms].q[1] = *(const uint4*)(&sA[m * 32 + kb + 16]);
    }
#pragma unroll
    for (int ns = 0; ns < 2; ++ns) {
      int n  = wn * 32 + ns * 16 + lo;
      int kb = hi * 8;  // low lanes K 0-7 & 16-23, high lanes 8-15 & 24-31
      b[ns].q[0] = *(const uint4*)(&sB[n * 32 + kb]);
      b[ns].q[1] = *(const uint4*)(&sB[n * 32 + kb + 16]);
    }
#pragma unroll
    for (int ms = 0; ms < 2; ++ms)
#pragma unroll
      for (int ns = 0; ns < 2; ++ns)
        acc[ms][ns] = wmma_bf16(a[ms].v, b[ns].v, acc[ms][ns]);
    __syncthreads();
  }

  // ---- epilogue: alpha, bias, relu, store (f32 | bf16 | bf16 transposed) ----
#pragma unroll
  for (int ms = 0; ms < 2; ++ms) {
#pragma unroll
    for (int ns = 0; ns < 2; ++ns) {
      int n = bn + wn * 32 + ns * 16 + lo;
      float bv = bias ? bias[n] : 0.0f;
#pragma unroll
      for (int r = 0; r < 8; ++r) {
        int m = bm + wm * 32 + ms * 16 + r + hi * 8;
        float v = acc[ms][ns][r] * alpha + bv;
        if (RELU) v = v > 0.0f ? v : 0.0f;
        if constexpr (OUT_TRANS) {
          ((__bf16*)Cp)[cOff + (long long)n * ldc + m] = f2bf(v);
        } else if constexpr (OUT_BF16) {
          ((__bf16*)Cp)[cOff + (long long)m * ldc + n] = f2bf(v);
        } else {
          ((float*)Cp)[cOff + (long long)m * ldc + n] = v;
        }
      }
    }
  }
}

// ------------------------- row softmax (in place) --------------------------

__global__ __launch_bounds__(256) void k_softmax(float* __restrict__ data, int ncol) {
  __shared__ float red[256];
  size_t base = (size_t)blockIdx.x * ncol;
  int t = threadIdx.x;
  float mx = -3.0e38f;
  for (int c = t; c < ncol; c += 256) mx = fmaxf(mx, data[base + c]);
  red[t] = mx; __syncthreads();
  for (int s = 128; s > 0; s >>= 1) { if (t < s) red[t] = fmaxf(red[t], red[t + s]); __syncthreads(); }
  mx = red[0]; __syncthreads();
  float sum = 0.0f;
  for (int c = t; c < ncol; c += 256) {
    float e = __expf(data[base + c] - mx);
    data[base + c] = e;
    sum += e;
  }
  red[t] = sum; __syncthreads();
  for (int s = 128; s > 0; s >>= 1) { if (t < s) red[t] += red[t + s]; __syncthreads(); }
  float inv = 1.0f / red[0];
  for (int c = t; c < ncol; c += 256) data[base + c] *= inv;
}

// -------------------- residual + LayerNorm (+add, +bf16) -------------------
// out = LN(a + b) * g + be  [+ extra];  optionally mirror to bf16

__global__ __launch_bounds__(256) void k_add_ln(
    const float* __restrict__ a, const float* __restrict__ b,
    const float* __restrict__ g, const float* __restrict__ be,
    float* __restrict__ out, const float* __restrict__ extra,
    __bf16* __restrict__ outBf)
{
  __shared__ float red[256];
  __shared__ float stat[2];
  int row = blockIdx.x, t = threadIdx.x;
  size_t base = (size_t)row * D_;
  float x[4];
  float s = 0.0f;
#pragma unroll
  for (int i = 0; i < 4; ++i) {
    int c = t + i * 256;
    x[i] = a[base + c] + b[base + c];
    s += x[i];
  }
  red[t] = s; __syncthreads();
  for (int k = 128; k > 0; k >>= 1) { if (t < k) red[t] += red[t + k]; __syncthreads(); }
  if (t == 0) stat[0] = red[0] * (1.0f / D_);
  __syncthreads();
  float mean = stat[0];
  float vs = 0.0f;
#pragma unroll
  for (int i = 0; i < 4; ++i) { float d = x[i] - mean; vs += d * d; }
  red[t] = vs; __syncthreads();
  for (int k = 128; k > 0; k >>= 1) { if (t < k) red[t] += red[t + k]; __syncthreads(); }
  if (t == 0) stat[1] = rsqrtf(red[0] * (1.0f / D_) + 1e-5f);
  __syncthreads();
  float inv = stat[1];
#pragma unroll
  for (int i = 0; i < 4; ++i) {
    int c = t + i * 256;
    float y = (x[i] - mean) * inv * g[c] + be[c];
    if (extra) y += extra[base + c];
    out[base + c] = y;
    if (outBf) outBf[base + c] = f2bf(y);
  }
}

// --------------------- fused banded (sliding window) attention -------------
// One wave per 16-query tile. Fixed 18 chunks of 16 keys (288) covering
// [q0-127, q0+142] for every tile -> uniform loops, safe block barriers.

#define NK16 18
#define LW   (NK16 * 16)   // 288

__global__ __launch_bounds__(128) void k_band_attn(
    const __bf16* __restrict__ q2, const __bf16* __restrict__ k2,
    const __bf16* __restrict__ v2t, __bf16* __restrict__ ctx2)
{
  __shared__ __align__(16) float sc[4][16][LW];  // 72 KB
  __shared__ float red[4][2][16];

  int t    = threadIdx.x;
  int lane = t & 31;
  int w    = t >> 5;
  int lo   = lane & 15;
  int hi   = lane >> 4;

  int idx = blockIdx.x * 4 + w;        // 0..4095 q-tiles
  int qt  = idx & (S_ / 16 - 1);
  int h   = (idx >> 7) & (H_ - 1);
  int b   = idx >> 11;
  int q0  = qt * 16;
  int ks  = q0 - (WIN_ - 1); if (ks < 0) ks = 0; ks &= ~15;

  // Q fragments (K=64 -> two 16x32 A-frags)
  FragAB a0, a1;
  {
    long long rbase = (long long)(b * S_ + q0 + lo) * (H_ * DK_) + h * DK_;
    int kb = hi * 8;
    a0.q[0] = *(const uint4*)(q2 + rbase + kb);
    a0.q[1] = *(const uint4*)(q2 + rbase + kb + 16);
    a1.q[0] = *(const uint4*)(q2 + rbase + 32 + kb);
    a1.q[1] = *(const uint4*)(q2 + rbase + 32 + kb + 16);
  }

  // ---- scores into LDS (masked, scaled) ----
  for (int kc = 0; kc < NK16; ++kc) {
    int key  = ks + kc * 16 + lo;
    int krow = key < S_ ? key : S_ - 1;     // clamp; masked below anyway
    long long rbase = (long long)(b * S_ + krow) * (H_ * DK_) + h * DK_;
    int kb2 = hi * 8;
    FragAB b0, b1;
    b0.q[0] = *(const uint4*)(k2 + rbase + kb2);
    b0.q[1] = *(const uint4*)(k2 + rbase + kb2 + 16);
    b1.q[0] = *(const uint4*)(k2 + rbase + 32 + kb2);
    b1.q[1] = *(const uint4*)(k2 + rbase + 32 + kb2 + 16);
    v8f c = {};
    c = wmma_bf16(a0.v, b0.v, c);
    c = wmma_bf16(a1.v, b1.v, c);
#pragma unroll
    for (int r = 0; r < 8; ++r) {
      int m = r + hi * 8;
      int q = q0 + m;
      int k = ks + kc * 16 + lo;
      int d = q - k; if (d < 0) d = -d;
      bool ok = (k < S_) && (d < WIN_);
      sc[w][m][kc * 16 + lo] = ok ? c[r] * 0.125f : -3.0e38f;
    }
  }
  __syncthreads();

  // ---- softmax over each of the 16 rows; lane pair (lo, lo+16) splits row ----
  {
    int m  = lo;
    int c0 = hi * (LW / 2);
    float mx = -3.0e38f;
    for (int c = 0; c < LW / 2; ++c) mx = fmaxf(mx, sc[w][m][c0 + c]);
    red[w][hi][m] = mx;
    __syncthreads();
    mx = fmaxf(red[w][0][m], red[w][1][m]);
    __syncthreads();
    float sum = 0.0f;
    for (int c = 0; c < LW / 2; ++c) {
      float e = __expf(sc[w][m][c0 + c] - mx);
      sc[w][m][c0 + c] = e;
      sum += e;
    }
    red[w][hi][m] = sum;
    __syncthreads();
    float inv = 1.0f / (red[w][0][m] + red[w][1][m]);
    for (int c = 0; c < LW / 2; ++c) sc[w][m][c0 + c] *= inv;
  }
  __syncthreads();

  // ---- P @ V : 9 chunks of K=32 keys, N=64 via 4 WMMA tiles ----
  v8f acc[4] = {};
  for (int kc = 0; kc < NK16 / 2; ++kc) {
    FragAB af;
    {
      int m = lo, kb = hi * 8;
#pragma unroll
      for (int j = 0; j < 8; ++j) af.e[j]     = f2bf(sc[w][m][kc * 32 + kb + j]);
#pragma unroll
      for (int j = 0; j < 8; ++j) af.e[8 + j] = f2bf(sc[w][m][kc * 32 + kb + 16 + j]);
    }
#pragma unroll
    for (int d = 0; d < 4; ++d) {
      long long vbase = (long long)(h * DV_ + d * 16 + lo) * (long long)NTOK_;
      int col = ks + kc * 32 + hi * 8;
      if (col >= S_) col = S_ - 16;  // tail: probs there are exactly 0
      FragAB bf_;
      bf_.q[0] = *(const uint4*)(v2t + vbase + b * S_ + col);
      bf_.q[1] = *(const uint4*)(v2t + vbase + b * S_ + col + 16);
      acc[d] = wmma_bf16(af.v, bf_.v, acc[d]);
    }
  }

  // ---- store ctx2 (bf16, [NTOK][H*DV]) ----
#pragma unroll
  for (int d = 0; d < 4; ++d)
#pragma unroll
    for (int r = 0; r < 8; ++r) {
      int m = r + hi * 8;
      long long o = (long long)(b * S_ + q0 + m) * (H_ * DV_) + h * DV_ + d * 16 + lo;
      ctx2[o] = f2bf(acc[d][r]);
    }
}

// ------------------------------ host driver --------------------------------

extern "C" void kernel_launch(void* const* d_in, const int* in_sizes, int n_in,
                              void* d_out, int out_size, void* d_ws, size_t ws_size,
                              hipStream_t stream) {
  (void)in_sizes; (void)n_in; (void)out_size; (void)ws_size;

  const float* x   = (const float*)d_in[0];
  const float* wq1 = (const float*)d_in[1];  const float* bq1 = (const float*)d_in[2];
  const float* wk1 = (const float*)d_in[3];  const float* bk1 = (const float*)d_in[4];
  const float* wv1 = (const float*)d_in[5];  const float* bv1 = (const float*)d_in[6];
  const float* wo1 = (const float*)d_in[7];  const float* bo1 = (const float*)d_in[8];
  const float* g1  = (const float*)d_in[9];  const float* be1 = (const float*)d_in[10];
  const float* wq2 = (const float*)d_in[11]; const float* bq2 = (const float*)d_in[12];
  const float* wk2 = (const float*)d_in[13]; const float* bk2 = (const float*)d_in[14];
  const float* wv2 = (const float*)d_in[15]; const float* bv2 = (const float*)d_in[16];
  const float* wo2 = (const float*)d_in[17]; const float* bo2 = (const float*)d_in[18];
  const float* g2  = (const float*)d_in[19]; const float* be2 = (const float*)d_in[20];
  const float* wc1 = (const float*)d_in[21]; const float* bc1 = (const float*)d_in[22];
  const float* wc2 = (const float*)d_in[23]; const float* bc2 = (const float*)d_in[24];
  const float* g3  = (const float*)d_in[25]; const float* be3 = (const float*)d_in[26];

  float* enc  = (float*)d_out;
  float* attn = enc + (size_t)NTOK_ * D_;   // [B,H,S,S]

  char* W = (char*)d_ws;
  size_t off = 0;
  auto alloc = [&](size_t bytes) -> void* {
    void* p = (void*)(W + off);
    off = (off + bytes + 255) & ~(size_t)255;
    return p;
  };

  const size_t TD   = (size_t)NTOK_ * D_;        // 4096*1024
  __bf16* xb    = (__bf16*)alloc(TD * 2);
  __bf16* wq1T  = (__bf16*)alloc((size_t)D_ * D_ * 2);
  __bf16* wk1T  = (__bf16*)alloc((size_t)D_ * D_ * 2);
  __bf16* wv1T  = (__bf16*)alloc((size_t)D_ * D_ * 2);
  __bf16* wo1T  = (__bf16*)alloc((size_t)D_ * D_ * 2);
  __bf16* wq2T  = (__bf16*)alloc((size_t)D_ * D_ * 2);
  __bf16* wk2T  = (__bf16*)alloc((size_t)D_ * D_ * 2);
  __bf16* wv2T  = (__bf16*)alloc((size_t)D_ * D_ * 2);
  __bf16* wo2T  = (__bf16*)alloc((size_t)D_ * D_ * 2);
  __bf16* wc1T  = (__bf16*)alloc((size_t)D_ * DFF_ * 2);
  __bf16* wc2T  = (__bf16*)alloc((size_t)D_ * DFF_ * 2);
  __bf16* q1b   = (__bf16*)alloc(TD * 2);
  __bf16* k1b   = (__bf16*)alloc(TD * 2);
  __bf16* q2b   = (__bf16*)alloc(TD * 2);
  __bf16* k2b   = (__bf16*)alloc(TD * 2);
  __bf16* v1t   = (__bf16*)alloc(TD * 2);   // [D][NTOK] transposed
  __bf16* v2t   = (__bf16*)alloc(TD * 2);
  __bf16* ctx1  = (__bf16*)alloc(TD * 2);
  __bf16* ctx2  = (__bf16*)alloc(TD * 2);
  __bf16* sb    = (__bf16*)alloc(TD * 2);
  __bf16* hb    = (__bf16*)alloc((size_t)NTOK_ * DFF_ * 2);
  float*  t1    = (float*)alloc(TD * 4);    // reused as ffn2 output
  float*  t2    = (float*)alloc(TD * 4);    // reused as s (in-place LN)
  float*  o1    = (float*)alloc(TD * 4);
  float*  s     = t2;
  float*  ffo   = t1;

  const int TPB = 256;
  const long long Z0 = 0;
  const long long SS   = (long long)S_ * S_;
  const long long HSS  = (long long)H_ * SS;
  const long long BTOK = (long long)S_ * D_;  // per-batch token stride in q/k/v

  // ---- converts ----
  k_cvt_bf16<<<(int)((TD + TPB - 1) / TPB), TPB, 0, stream>>>(x, xb, (int)TD);
  int wn2 = D_ * D_;
  k_cvt_bf16_T<<<(wn2 + TPB - 1) / TPB, TPB, 0, stream>>>(wq1, wq1T, D_, D_);
  k_cvt_bf16_T<<<(wn2 + TPB - 1) / TPB, TPB, 0, stream>>>(wk1, wk1T, D_, D_);
  k_cvt_bf16_T<<<(wn2 + TPB - 1) / TPB, TPB, 0, stream>>>(wv1, wv1T, D_, D_);
  k_cvt_bf16_T<<<(wn2 + TPB - 1) / TPB, TPB, 0, stream>>>(wo1, wo1T, D_, D_);
  k_cvt_bf16_T<<<(wn2 + TPB - 1) / TPB, TPB, 0, stream>>>(wq2, wq2T, D_, D_);
  k_cvt_bf16_T<<<(wn2 + TPB - 1) / TPB, TPB, 0, stream>>>(wk2, wk2T, D_, D_);
  k_cvt_bf16_T<<<(wn2 + TPB - 1) / TPB, TPB, 0, stream>>>(wv2, wv2T, D_, D_);
  k_cvt_bf16_T<<<(wn2 + TPB - 1) / TPB, TPB, 0, stream>>>(wo2, wo2T, D_, D_);
  int wnf = D_ * DFF_;
  k_cvt_bf16_T<<<(wnf + TPB - 1) / TPB, TPB, 0, stream>>>(wc1, wc1T, D_, DFF_);
  k_cvt_bf16_T<<<(wnf + TPB - 1) / TPB, TPB, 0, stream>>>(wc2, wc2T, DFF_, D_);

  // ---- projections (M=4096, N=1024, K=1024) ----
  dim3 gProj(D_ / 64, NTOK_ / 128, 1);
  k_gemm<__bf16, true,  false, false><<<gProj, TPB, 0, stream>>>(
      xb, wq1T, bq1, q1b, NTOK_, D_, D_, D_, D_, D_, Z0,Z0,Z0,Z0,Z0,Z0, 1, 1.0f);
  k_gemm<__bf16, true,  false, false><<<gProj, TPB, 0, stream>>>(
      xb, wk1T, bk1, k1b, NTOK_, D_, D_, D_, D_, D_, Z0,Z0,Z0,Z0,Z0,Z0, 1, 1.0f);
  k_gemm<__bf16, true,  true,  false><<<gProj, TPB, 0, stream>>>(
      xb, wv1T, bv1, v1t, NTOK_, D_, D_, D_, D_, NTOK_, Z0,Z0,Z0,Z0,Z0,Z0, 1, 1.0f);
  k_gemm<__bf16, true,  false, false><<<gProj, TPB, 0, stream>>>(
      xb, wq2T, bq2, q2b, NTOK_, D_, D_, D_, D_, D_, Z0,Z0,Z0,Z0,Z0,Z0, 1, 1.0f);
  k_gemm<__bf16, true,  false, false><<<gProj, TPB, 0, stream>>>(
      xb, wk2T, bk2, k2b, NTOK_, D_, D_, D_, D_, D_, Z0,Z0,Z0,Z0,Z0,Z0, 1, 1.0f);
  k_gemm<__bf16, true,  true,  false><<<gProj, TPB, 0, stream>>>(
      xb, wv2T, bv2, v2t, NTOK_, D_, D_, D_, D_, NTOK_, Z0,Z0,Z0,Z0,Z0,Z0, 1, 1.0f);

  // ---- full attention scores -> attn output region (alpha = 1/sqrt(64)) ----
  dim3 gScore(S_ / 64, S_ / 128, BB_ * H_);
  k_gemm<__bf16, false, false, false><<<gScore, TPB, 0, stream>>>(
      q1b, k1b, nullptr, attn, S_, S_, DK_, D_, D_, S_,
      BTOK, (long long)DK_, BTOK, (long long)DK_, HSS, SS, H_, 0.125f);

  // ---- softmax rows (B*H*S rows of S) ----
  k_softmax<<<BB_ * H_ * S_, TPB, 0, stream>>>(attn, S_);

  // ---- ctx1 = attn @ V  (A is fp32, converted while staging) ----
  dim3 gCtx(DV_ / 64, S_ / 128, BB_ * H_);
  k_gemm<float, true, false, false><<<gCtx, TPB, 0, stream>>>(
      attn, v1t, nullptr, ctx1, S_, DV_, S_, S_, NTOK_, D_,
      HSS, SS, (long long)S_, (long long)DV_ * NTOK_, BTOK, (long long)DV_, H_, 1.0f);

  // ---- sliding window attention (fused) ----
  k_band_attn<<<(BB_ * H_ * (S_ / 16)) / 4, 128, 0, stream>>>(q2b, k2b, v2t, ctx2);

  // ---- output projections ----
  k_gemm<__bf16, false, false, false><<<gProj, TPB, 0, stream>>>(
      ctx1, wo1T, bo1, t1, NTOK_, D_, D_, D_, D_, D_, Z0,Z0,Z0,Z0,Z0,Z0, 1, 1.0f);
  k_gemm<__bf16, false, false, false><<<gProj, TPB, 0, stream>>>(
      ctx2, wo2T, bo2, t2, NTOK_, D_, D_, D_, D_, D_, Z0,Z0,Z0,Z0,Z0,Z0, 1, 1.0f);

  // ---- o1 = LN(t1 + x); s = LN(t2 + x) + o1 (fp32 + bf16 mirror) ----
  k_add_ln<<<NTOK_, TPB, 0, stream>>>(t1, x, g1, be1, o1, nullptr, nullptr);
  k_add_ln<<<NTOK_, TPB, 0, stream>>>(t2, x, g2, be2, s, o1, sb);

  // ---- FFN ----
  dim3 gF1(DFF_ / 64, NTOK_ / 128, 1);
  k_gemm<__bf16, true, false, true><<<gF1, TPB, 0, stream>>>(
      sb, wc1T, bc1, hb, NTOK_, DFF_, D_, D_, D_, DFF_, Z0,Z0,Z0,Z0,Z0,Z0, 1, 1.0f);
  dim3 gF2(D_ / 64, NTOK_ / 128, 1);
  k_gemm<__bf16, false, false, false><<<gF2, TPB, 0, stream>>>(
      hb, wc2T, bc2, ffo, NTOK_, D_, DFF_, DFF_, DFF_, D_, Z0,Z0,Z0,Z0,Z0,Z0, 1, 1.0f);

  // ---- enc = LN(ffo + s) ----
  k_add_ln<<<NTOK_, TPB, 0, stream>>>(ffo, s, g3, be3, enc, nullptr, nullptr);
}
